// TextConditionedDynamicLayerAttention_82471962018682
// MI455X (gfx1250) — matrix operand order repaired and checked by reading, not compiled, not run on hardware
//
#include <hip/hip_runtime.h>

// ---------------- problem constants ----------------
constexpr int CL   = 24;          // layers
constexpr int CN   = 576;         // tokens per layer
constexpr int CD   = 4096;        // dim
constexpr int CT   = 512;         // text tokens
constexpr int CRED = 1024;        // D/4
constexpr int CTOK = CL * CN;     // 13824
constexpr int FOCUS = 22;
#define EPS_LN  1e-5f
#define EPS_STD 1e-6f
#define NEGV   (-1e30f)

// ---------------- CDNA5 feature probes ----------------
#if defined(__has_builtin)
# if __has_builtin(__builtin_amdgcn_global_load_async_to_lds_b128) && \
     __has_builtin(__builtin_amdgcn_s_wait_asynccnt)
#  define USE_ASYNC_LDS 1
# endif
#endif
#ifndef USE_ASYNC_LDS
# define USE_ASYNC_LDS 0
#endif

#if USE_ASYNC_LDS
// builtin signature (verified via hipcc diagnostic): v4i* in global / LDS
// address spaces, then two immediate ints (offset, cpol).
typedef __attribute__((ext_vector_type(4))) int v4i;
typedef __attribute__((address_space(1))) v4i gv4i;
typedef __attribute__((address_space(3))) v4i lv4i;
#endif

// ---------------- WMMA types (CDNA5, wave32) ----------------
typedef __attribute__((ext_vector_type(16))) __bf16 v16bf;
typedef __attribute__((ext_vector_type(8)))  float  v8f;

__device__ inline unsigned short f2bfbits(float f) {
  unsigned u = __builtin_bit_cast(unsigned, f);
  u += 0x7FFFu + ((u >> 16) & 1u);            // round-to-nearest-even
  return (unsigned short)(u >> 16);
}
__device__ inline __bf16 f2bf(float f) {
  return __builtin_bit_cast(__bf16, f2bfbits(f));
}

__device__ inline float waveRedSum(float v) {
  for (int m = 16; m >= 1; m >>= 1) v += __shfl_xor(v, m, 32);
  return v;
}

// ---------------- small prep kernels ----------------
__global__ void k_text_mean(const float* __restrict__ tf, float* __restrict__ tmean) {
  int d = blockIdx.x * 256 + threadIdx.x;
  float s = 0.f;
  for (int t = 0; t < CT; ++t) s += tf[(size_t)t * CD + d];
  tmean[d] = s * (1.0f / CT);
}

__global__ void k_ln_noaffine(const float* __restrict__ x, float* __restrict__ yv) {
  __shared__ float r1[256], r2[256];
  int tid = threadIdx.x;
  float s = 0.f, q = 0.f;
  for (int i = tid; i < CD; i += 256) { float v = x[i]; s += v; q += v * v; }
  r1[tid] = s; r2[tid] = q; __syncthreads();
  for (int st = 128; st > 0; st >>= 1) {
    if (tid < st) { r1[tid] += r1[tid + st]; r2[tid] += r2[tid + st]; }
    __syncthreads();
  }
  float m  = r1[0] * (1.0f / CD);
  float va = r2[0] * (1.0f / CD) - m * m;
  float rs = rsqrtf(va + EPS_LN);
  for (int i = tid; i < CD; i += 256) yv[i] = (x[i] - m) * rs;
}

__global__ void k_layer_mean(const float* __restrict__ lf, float* __restrict__ y) {
  int l = blockIdx.y;
  int d = blockIdx.x * 256 + threadIdx.x;
  const float* base = lf + (size_t)l * CN * CD + d;
  float s = 0.f;
  for (int n = 0; n < CN; ++n) s += base[(size_t)n * CD];
  y[l * CD + d] = s * (1.0f / CN);
}

// f32 -> packed bf16 (2 per u32), streaming one-pass conversion
__global__ void k_cvt_bf16(const float* __restrict__ src, unsigned int* __restrict__ dst,
                           int n2) {
  int i = blockIdx.x * 256 + threadIdx.x;
  if (i < n2) {
    unsigned lo = f2bfbits(src[2 * i]);
    unsigned hi = f2bfbits(src[2 * i + 1]);
    dst[i] = lo | (hi << 16);
  }
}

// ---------------- DSU recurrence: one persistent workgroup ----------------
__global__ void __launch_bounds__(1024)
k_dsu(const float* __restrict__ y, const float* __restrict__ tg,
      const float* __restrict__ W1w, const float* __restrict__ W1b,
      const float* __restrict__ Wcw, const float* __restrict__ Wcb,
      const float* __restrict__ Wiw, const float* __restrict__ Wib,
      const float* __restrict__ Wfw, const float* __restrict__ Wfb,
      const float* __restrict__ bc, const float* __restrict__ bi,
      const float* __restrict__ bf, float* __restrict__ cfin) {
  __shared__ float comb[3 * CD];   // 48 KB
  __shared__ float sarr[CRED];     //  4 KB
  __shared__ float cbuf[CD];       // 16 KB
  int tid = threadIdx.x;
  for (int i = tid; i < CD; i += 1024) cbuf[i] = 0.f;
  __syncthreads();
  for (int l = 0; l <= FOCUS; ++l) {          // contexts[L-2] = after step 22
    for (int i = tid; i < CD; i += 1024) {
      float cv = cbuf[i];
      comb[i]          = 1.f / (1.f + expf(-cv));
      comb[CD + i]     = y[l * CD + i];
      comb[2 * CD + i] = tg[i];
    }
    __syncthreads();
    {
      const float* wrow = W1w + (size_t)tid * (3 * CD);
      float acc = 0.f;
      for (int i = 0; i < 3 * CD; ++i) acc += wrow[i] * comb[i];
      acc += W1b[tid];
      sarr[tid] = acc > 0.f ? acc : 0.f;
    }
    __syncthreads();
    float cn[4];
    for (int k4 = 0; k4 < 4; ++k4) {
      int d = tid + k4 * 1024;
      const float* rc = Wcw + (size_t)d * CRED;
      const float* ri = Wiw + (size_t)d * CRED;
      const float* rf = Wfw + (size_t)d * CRED;
      float ac = 0.f, ai = 0.f, af = 0.f;
      for (int j = 0; j < CRED; ++j) {
        float sv = sarr[j];
        ac += rc[j] * sv; ai += ri[j] * sv; af += rf[j] * sv;
      }
      float ct = tanhf(ac + Wcb[d] + bc[d]);
      float ig = 1.f / (1.f + expf(-(ai + Wib[d] + bi[d])));
      float fg = 1.f / (1.f + expf(-(af + Wfb[d] + bf[d])));
      cn[k4] = fg * cbuf[d] + ig * ct;
    }
    __syncthreads();
    for (int k4 = 0; k4 < 4; ++k4) cbuf[tid + k4 * 1024] = cn[k4];
    __syncthreads();
  }
  for (int i = tid; i < CD; i += 1024) cfin[i] = cbuf[i];
}

// ---------------- q projection + normalization ----------------
__global__ void k_qraw(const float* __restrict__ qw, const float* __restrict__ cfin,
                       float* __restrict__ qraw) {
  int wave = threadIdx.x >> 5, lane = threadIdx.x & 31;
  int d = blockIdx.x * 8 + wave;
  const float* row = qw + (size_t)d * CD;
  float acc = 0.f;
  for (int e = lane; e < CD; e += 32) acc += row[e] * cfin[e];
  acc = waveRedSum(acc);
  if (lane == 0) qraw[d] = acc;
}

__global__ void k_qnorm(const float* __restrict__ qraw, const float* __restrict__ g,
                        const float* __restrict__ b, float* __restrict__ gq,
                        float* __restrict__ scal) {
  __shared__ float r1[256], r2[256];
  int tid = threadIdx.x;
  float s = 0.f, q = 0.f;
  for (int i = tid; i < CD; i += 256) { float v = qraw[i]; s += v; q += v * v; }
  r1[tid] = s; r2[tid] = q; __syncthreads();
  for (int st = 128; st > 0; st >>= 1) {
    if (tid < st) { r1[tid] += r1[tid + st]; r2[tid] += r2[tid + st]; }
    __syncthreads();
  }
  float m  = r1[0] * (1.0f / CD);
  float va = r2[0] * (1.0f / CD) - m * m;
  float rs = rsqrtf(va + EPS_LN);
  __syncthreads();
  float sgq = 0.f, sbq = 0.f;
  for (int i = tid; i < CD; i += 256) {
    float qv = (qraw[i] - m) * rs * g[i] + b[i];
    float gv = g[i] * qv;
    gq[i] = gv; sgq += gv; sbq += b[i] * qv;
  }
  r1[tid] = sgq; r2[tid] = sbq; __syncthreads();
  for (int st = 128; st > 0; st >>= 1) {
    if (tid < st) { r1[tid] += r1[tid + st]; r2[tid] += r2[tid + st]; }
    __syncthreads();
  }
  if (tid == 0) { scal[0] = r1[0]; scal[1] = r2[0]; }
}

// u[e] = colmean of k_w ; wv[e] = sum_d k_w[d,e]*gq[d]
__global__ void k_uwv(const float* __restrict__ kw, const float* __restrict__ gq,
                      float* __restrict__ u, float* __restrict__ wv) {
  int e = blockIdx.x * 256 + threadIdx.x;
  float su = 0.f, sv = 0.f;
  for (int d = 0; d < CD; ++d) {
    float kv = kw[(size_t)d * CD + e];
    su += kv; sv += kv * gq[d];
  }
  u[e] = su * (1.0f / CD);
  wv[e] = sv;
}

__global__ void k_zero(float* __restrict__ p, int n) {
  int i = blockIdx.x * 256 + threadIdx.x;
  if (i < n) p[i] = 0.f;
}

// ---------------- fused GEMM + row sum-of-squares (WMMA bf16) ----------------
// sumsq[r] = sum_d ( sum_e A[r,e]*kw[d,e] )^2 ; A = layer_feats [13824 x 4096]
constexpr int MB = 64, NB = 128, KB = 32;

// Shared epilogue: C layout -> VGPR j = row j (lanes 0-15) / j+8 (lanes 16-31).
#define SUMSQ_EPILOGUE                                                      \
  _Pragma("unroll")                                                         \
  for (int mi = 0; mi < 2; ++mi) {                                          \
    _Pragma("unroll")                                                       \
    for (int j = 0; j < 8; ++j) {                                           \
      float v0 = acc[mi][0][j];                                             \
      float v1 = acc[mi][1][j];                                             \
      float s = v0 * v0 + v1 * v1;                                          \
      s += __shfl_xor(s, 1, 32);                                            \
      s += __shfl_xor(s, 2, 32);                                            \
      s += __shfl_xor(s, 4, 32);                                            \
      s += __shfl_xor(s, 8, 32);                                            \
      if ((lane & 15) == 0) {                                               \
        int row = m0 + wr * 32 + mi * 16 + j + ((lane >= 16) ? 8 : 0);      \
        atomicAdd(&sumsq[row], s);                                          \
      }                                                                     \
    }                                                                       \
  }

// Shared fragment-build + 4x WMMA for one K-chunk (reads As/Bs of __bf16).
#define WMMA_CHUNK(ASRC, BSRC)                                              \
  {                                                                         \
    const int half = lane >> 4;                                             \
    const int l16  = lane & 15;                                             \
    v16bf afrag[2], bfrag[2];                                               \
    _Pragma("unroll")                                                       \
    for (int mi = 0; mi < 2; ++mi) {                                        \
      const int r = wr * 32 + mi * 16 + l16;                                \
      _Pragma("unroll")                                                     \
      for (int j = 0; j < 8; ++j) {                                         \
        afrag[mi][j]     = ASRC[r][half * 8 + j];                           \
        afrag[mi][8 + j] = ASRC[r][16 + half * 8 + j];                      \
      }                                                                     \
    }                                                                       \
    _Pragma("unroll")                                                       \
    for (int ni = 0; ni < 2; ++ni) {                                        \
      const int c = wc * 32 + ni * 16 + l16;                                \
      _Pragma("unroll")                                                     \
      for (int j = 0; j < 8; ++j) {                                         \
        bfrag[ni][j]     = BSRC[c][half * 8 + j];                           \
        bfrag[ni][8 + j] = BSRC[c][16 + half * 8 + j];                      \
      }                                                                     \
    }                                                                       \
    _Pragma("unroll")                                                       \
    for (int mi = 0; mi < 2; ++mi)                                          \
      _Pragma("unroll")                                                     \
      for (int ni = 0; ni < 2; ++ni)                                        \
        acc[mi][ni] = __builtin_amdgcn_wmma_f32_16x16x32_bf16(              \
            false, afrag[mi], false, bfrag[ni], (short)0, acc[mi][ni],      \
            false, false);                                                  \
  }

// Fallback: f32 inputs, convert-at-store into bf16 LDS.
__global__ void __launch_bounds__(256)
k_gemm_sumsq_sync(const float* __restrict__ A, const float* __restrict__ Bw,
                  float* __restrict__ sumsq) {
  __shared__ __bf16 As[MB][KB + 2];
  __shared__ __bf16 Bs[NB][KB + 2];
  const int tid  = threadIdx.x;
  const int lane = tid & 31;
  const int wave = tid >> 5;
  const int wr = wave >> 2;
  const int wc = wave & 3;
  const int m0 = blockIdx.x * MB;
  const int n0 = blockIdx.y * NB;

  v8f zero = {};
  v8f acc[2][2];
  acc[0][0] = zero; acc[0][1] = zero; acc[1][0] = zero; acc[1][1] = zero;

  const int ar  = tid >> 2;
  const int ac0 = (tid & 3) * 8;
  const int br  = tid >> 1;
  const int bc0 = (tid & 1) * 16;

  for (int kb = 0; kb < CD; kb += KB) {
    if (kb + KB < CD) {
      __builtin_prefetch(&A [(size_t)(m0 + ar) * CD + kb + KB + ac0], 0, 1);
      __builtin_prefetch(&Bw[(size_t)(n0 + br) * CD + kb + KB + bc0], 0, 1);
    }
    {
      const float* src = &A[(size_t)(m0 + ar) * CD + kb + ac0];
      #pragma unroll
      for (int i = 0; i < 8; ++i) As[ar][ac0 + i] = f2bf(src[i]);
    }
    {
      const float* src = &Bw[(size_t)(n0 + br) * CD + kb + bc0];
      #pragma unroll
      for (int i = 0; i < 16; ++i) Bs[br][bc0 + i] = f2bf(src[i]);
    }
    __syncthreads();
    WMMA_CHUNK(As, Bs)
    __syncthreads();
  }
  SUMSQ_EPILOGUE
}

#if USE_ASYNC_LDS
// Preferred: pre-converted bf16 matrices, tiles DMA'd by the async
// global->LDS engine; zero conversion work in the hot loop.
// Row pad = 8 bf16 (16 B): every b128 destination stays 16B-aligned and the
// 16 row addresses a half-wave reads hit distinct banks (20*r mod 64 distinct).
__global__ void __launch_bounds__(256)
k_gemm_sumsq_abf(const unsigned short* __restrict__ A,
                 const unsigned short* __restrict__ Bw,
                 float* __restrict__ sumsq) {
  __shared__ __bf16 As[MB][KB + 8];   //  5 KB
  __shared__ __bf16 Bs[NB][KB + 8];   // 10 KB
  const int tid  = threadIdx.x;
  const int lane = tid & 31;
  const int wave = tid >> 5;
  const int wr = wave >> 2;
  const int wc = wave & 3;
  const int m0 = blockIdx.x * MB;
  const int n0 = blockIdx.y * NB;

  v8f zero = {};
  v8f acc[2][2];
  acc[0][0] = zero; acc[0][1] = zero; acc[1][0] = zero; acc[1][1] = zero;

  const int arow = tid >> 2;          // A: 256 16B chunks, 1 per thread
  const int acol = (tid & 3) * 8;

  for (int kb = 0; kb < CD; kb += KB) {
    if (kb + KB < CD) {               // gfx1250 global_prefetch_b8
      __builtin_prefetch(&A [(size_t)(m0 + arow) * CD + kb + KB + acol], 0, 1);
      __builtin_prefetch(&Bw[(size_t)(n0 + (tid >> 1)) * CD + kb + KB], 0, 1);
    }
    {
      const unsigned short* src = &A[(size_t)(m0 + arow) * CD + kb + acol];
      __builtin_amdgcn_global_load_async_to_lds_b128(
          (gv4i*)(void*)src, (lv4i*)(void*)&As[arow][acol], 0, 0);
    }
    #pragma unroll
    for (int i = 0; i < 2; ++i) {     // B: 512 16B chunks, 2 per thread
      int q = tid * 2 + i;
      int row = q >> 2, c = (q & 3) * 8;
      const unsigned short* src = &Bw[(size_t)(n0 + row) * CD + kb + c];
      __builtin_amdgcn_global_load_async_to_lds_b128(
          (gv4i*)(void*)src, (lv4i*)(void*)&Bs[row][c], 0, 0);
    }
    __builtin_amdgcn_s_wait_asynccnt(0);
    __syncthreads();
    WMMA_CHUNK(As, Bs)
    __syncthreads();
  }
  SUMSQ_EPILOGUE
}
#endif  // USE_ASYNC_LDS

// per-token m and dot-score via the two reduction vectors
__global__ void k_tokstats(const float* __restrict__ A, const float* __restrict__ u,
                           const float* __restrict__ wv, float* __restrict__ mtok,
                           float* __restrict__ dstok) {
  int wave = threadIdx.x >> 5, lane = threadIdx.x & 31;
  int r = blockIdx.x * 8 + wave;
  const float* row = A + (size_t)r * CD;
  float am = 0.f, as = 0.f;
  for (int e = lane; e < CD; e += 32) {
    float f = row[e];
    am += f * u[e]; as += f * wv[e];
  }
  am = waveRedSum(am); as = waveRedSum(as);
  if (lane == 0) { mtok[r] = am; dstok[r] = as; }
}

// scores + per-layer conf
__global__ void k_scores(const float* __restrict__ sumsq, const float* __restrict__ mtok,
                         const float* __restrict__ dstok, const float* __restrict__ scal,
                         float* __restrict__ scores, float* __restrict__ conf) {
  __shared__ float r1[256], r2[256], r3[256];
  int l = blockIdx.x, tid = threadIdx.x;
  float Sgq = scal[0], Sbq = scal[1];
  float sum = 0.f, sq = 0.f, mx = -3.0e38f;
  for (int n = tid; n < CN; n += 256) {
    int r = l * CN + n;
    float mm = mtok[r];
    float va = sumsq[r] * (1.0f / CD) - mm * mm;
    if (va < 0.f) va = 0.f;
    float sc = rsqrtf(va + EPS_LN) * (dstok[r] - mm * Sgq) + Sbq;
    scores[r] = sc;
    sum += sc; sq += sc * sc; mx = fmaxf(mx, sc);
  }
  r1[tid] = sum; r2[tid] = sq; r3[tid] = mx; __syncthreads();
  for (int st = 128; st > 0; st >>= 1) {
    if (tid < st) {
      r1[tid] += r1[tid + st]; r2[tid] += r2[tid + st];
      r3[tid] = fmaxf(r3[tid], r3[tid + st]);
    }
    __syncthreads();
  }
  if (tid == 0) {
    float mean = r1[0] * (1.0f / CN);
    float va = r2[0] * (1.0f / CN) - mean * mean;
    if (va < 0.f) va = 0.f;
    conf[l] = (r3[0] - mean) / (sqrtf(va) + EPS_STD);
  }
}

// softmax allocation: floor + remainder by rank (single wave32)
__global__ void k_alloc(const float* __restrict__ conf, int* __restrict__ kplan) {
  __shared__ float w[23];
  int t = threadIdx.x;
  int idx = (t < FOCUS) ? t : 23;              // others = 0..21, 23
  bool act = t < 23;
  float cv = act ? conf[idx] : -3.0e38f;
  float mx = cv;
  for (int m = 16; m >= 1; m >>= 1) mx = fmaxf(mx, __shfl_xor(mx, m, 32));
  float ev = act ? expf(cv - mx) : 0.f;
  float s = ev;
  for (int m = 16; m >= 1; m >>= 1) s += __shfl_xor(s, m, 32);
  float wi = ev / s;
  int al = act ? (int)floorf(wi * 64.0f) : 0;
  int tot = al;
  for (int m = 16; m >= 1; m >>= 1) tot += __shfl_xor(tot, m, 32);
  int rem = 64 - tot;
  if (act) w[t] = wi;
  __syncthreads();
  if (act) {
    int rank = 0;
    for (int j = 0; j < 23; ++j)
      rank += (w[j] > wi) || (w[j] == wi && j < t);
    al += (rank < rem) ? 1 : 0;
    if (al > CN) al = CN;
    kplan[idx] = al;
  }
  if (t == 0) kplan[FOCUS] = 64;
}

// per-layer top-64 with lax.top_k tie semantics (lower index first)
__global__ void k_topk(const float* __restrict__ scores, const int* __restrict__ kplan,
                       float* __restrict__ topv, int* __restrict__ topi,
                       float* __restrict__ cand) {
  __shared__ float vals[CN];
  __shared__ float rv[256]; __shared__ int ri[256];
  int l = blockIdx.x, tid = threadIdx.x;
  for (int n = tid; n < CN; n += 256) vals[n] = scores[l * CN + n];
  __syncthreads();
  int kp = kplan[l];
  for (int it = 0; it < 64; ++it) {
    float bv = -3.4e38f; int bi = 1 << 30;
    for (int n = tid; n < CN; n += 256) {
      float v = vals[n];
      if (v > bv || (v == bv && n < bi)) { bv = v; bi = n; }
    }
    rv[tid] = bv; ri[tid] = bi; __syncthreads();
    for (int st = 128; st > 0; st >>= 1) {
      if (tid < st) {
        float v2 = rv[tid + st]; int i2 = ri[tid + st];
        if (v2 > rv[tid] || (v2 == rv[tid] && i2 < ri[tid])) { rv[tid] = v2; ri[tid] = i2; }
      }
      __syncthreads();
    }
    if (tid == 0) {
      topv[l * 64 + it] = rv[0];
      topi[l * 64 + it] = ri[0];
      cand[l * 64 + it] = (it < kp) ? rv[0] : NEGV;
      vals[ri[0]] = -3.4e38f;
    }
    __syncthreads();
  }
}

// focus keep 32 -> NEG them -> global keep 32 -> stable desc sort -> row indices
__global__ void k_select(const float* __restrict__ cand, const int* __restrict__ topi,
                         int* __restrict__ rowsel) {
  __shared__ float work[CL * 64];
  __shared__ float rv[256]; __shared__ int ri[256];
  __shared__ int   sel[64];
  __shared__ float selsc[64];
  int tid = threadIdx.x;
  for (int i = tid; i < CL * 64; i += 256) work[i] = cand[i];
  __syncthreads();
  // top-32 inside focus window [FOCUS*64, FOCUS*64+64)
  for (int it = 0; it < 32; ++it) {
    float bv = -3.4e38f; int bi = 1 << 30;
    if (tid < 64) {
      int g = FOCUS * 64 + tid;
      float v = work[g];
      bv = v; bi = g;
    }
    rv[tid] = bv; ri[tid] = bi; __syncthreads();
    for (int st = 128; st > 0; st >>= 1) {
      if (tid < st) {
        float v2 = rv[tid + st]; int i2 = ri[tid + st];
        if (v2 > rv[tid] || (v2 == rv[tid] && i2 < ri[tid])) { rv[tid] = v2; ri[tid] = i2; }
      }
      __syncthreads();
    }
    if (tid == 0) { sel[it] = ri[0]; work[ri[0]] = -3.4e38f; }
    __syncthreads();
  }
  // reference sets selected focus entries to NEG (still candidates for "rest")
  if (tid < 32) work[sel[tid]] = NEGV;
  __syncthreads();
  // global top-32 over all 1536
  for (int it = 32; it < 64; ++it) {
    float bv = -3.4e38f; int bi = 1 << 30;
    for (int i = tid; i < CL * 64; i += 256) {
      float v = work[i];
      if (v > bv || (v == bv && i < bi)) { bv = v; bi = i; }
    }
    rv[tid] = bv; ri[tid] = bi; __syncthreads();
    for (int st = 128; st > 0; st >>= 1) {
      if (tid < st) {
        float v2 = rv[tid + st]; int i2 = ri[tid + st];
        if (v2 > rv[tid] || (v2 == rv[tid] && i2 < ri[tid])) { rv[tid] = v2; ri[tid] = i2; }
      }
      __syncthreads();
    }
    if (tid == 0) { sel[it] = ri[0]; work[ri[0]] = -3.4e38f; }
    __syncthreads();
  }
  if (tid < 64) selsc[tid] = cand[sel[tid]];   // original (pre-mutation) scores
  __syncthreads();
  if (tid < 64) {
    float myv = selsc[tid];
    int rank = 0;
    for (int j = 0; j < 64; ++j) {
      float v = selsc[j];
      rank += (v > myv) || (v == myv && j < tid);
    }
    int s = sel[tid];
    int lay = s >> 6, pos = s & 63;
    int tok = topi[lay * 64 + pos];
    rowsel[rank] = lay * CN + tok;
  }
}

__global__ void k_gather(const float* __restrict__ lf, const int* __restrict__ rowsel,
                         float* __restrict__ out) {
  int b = blockIdx.x;
  int row = rowsel[b];
  const float* src = lf + (size_t)row * CD;
  float* dst = out + (size_t)b * CD;
  for (int i = threadIdx.x; i < CD; i += 256) dst[i] = src[i];
}

// ---------------- launcher ----------------
extern "C" void kernel_launch(void* const* d_in, const int* in_sizes, int n_in,
                              void* d_out, int out_size, void* d_ws, size_t ws_size,
                              hipStream_t stream) {
  const float* tf  = (const float*)d_in[0];
  const float* lf  = (const float*)d_in[1];
  const float* W1w = (const float*)d_in[2];
  const float* W1b = (const float*)d_in[3];
  const float* Wcw = (const float*)d_in[4];
  const float* Wcb = (const float*)d_in[5];
  const float* Wiw = (const float*)d_in[6];
  const float* Wib = (const float*)d_in[7];
  const float* Wfw = (const float*)d_in[8];
  const float* Wfb = (const float*)d_in[9];
  const float* bc  = (const float*)d_in[10];
  const float* bi  = (const float*)d_in[11];
  const float* bfv = (const float*)d_in[12];
  const float* qw  = (const float*)d_in[13];
  const float* kw  = (const float*)d_in[14];
  const float* lng = (const float*)d_in[15];
  const float* lnb = (const float*)d_in[16];
  float* out = (float*)d_out;
  float* ws  = (float*)d_ws;

  float* tmean  = ws + 0;
  float* tg     = ws + 4096;
  float* y      = ws + 8192;            // 24*4096
  float* cfin   = ws + 106496;
  float* qraw   = ws + 110592;
  float* gq     = ws + 114688;
  float* scal   = ws + 118784;          // [Sgq, Sbq]
  float* u      = ws + 118800;
  float* wv     = ws + 122896;
  float* sumsq  = ws + 126992;          // 13824
  float* mtok   = ws + 140816;
  float* dstok  = ws + 154640;
  float* scores = ws + 168464;
  float* conf   = ws + 182288;
  int*   kplan  = (int*)(ws + 182320);
  float* topv   = ws + 182352;
  int*   topi   = (int*)(ws + 183888);
  float* cand   = ws + 185424;
  int*   rowsel = (int*)(ws + 186960);

  k_text_mean <<<CD / 256, 256, 0, stream>>>(tf, tmean);
  k_ln_noaffine<<<1, 256, 0, stream>>>(tmean, tg);
  k_layer_mean<<<dim3(CD / 256, CL), 256, 0, stream>>>(lf, y);
  k_dsu       <<<1, 1024, 0, stream>>>(y, tg, W1w, W1b, Wcw, Wcb, Wiw, Wib,
                                       Wfw, Wfb, bc, bi, bfv, cfin);
  k_qraw      <<<CD / 8, 256, 0, stream>>>(qw, cfin, qraw);
  k_qnorm     <<<1, 256, 0, stream>>>(qraw, lng, lnb, gq, scal);
  k_uwv       <<<CD / 256, 256, 0, stream>>>(kw, gq, u, wv);
  k_zero      <<<(CTOK + 255) / 256, 256, 0, stream>>>(sumsq, CTOK);

#if USE_ASYNC_LDS
  // bf16 shadow copies of layer_feats and k_w live after the 1 MiB float region
  const size_t bfOff  = 262144;                       // floats (1 MiB)
  const size_t needB  = bfOff * 4 +
                        ((size_t)CTOK * CD + (size_t)CD * CD) * 2;
  if (ws_size >= needB) {
    unsigned short* abf = (unsigned short*)(ws + bfOff);
    unsigned short* kbf = abf + (size_t)CTOK * CD;
    const int nA2 = CTOK * CD / 2, nK2 = CD * CD / 2;
    k_cvt_bf16<<<(nA2 + 255) / 256, 256, 0, stream>>>(lf, (unsigned int*)abf, nA2);
    k_cvt_bf16<<<(nK2 + 255) / 256, 256, 0, stream>>>(kw, (unsigned int*)kbf, nK2);
    k_gemm_sumsq_abf<<<dim3(CTOK / MB, CD / NB), 256, 0, stream>>>(abf, kbf, sumsq);
  } else
#endif
  {
    k_gemm_sumsq_sync<<<dim3(CTOK / MB, CD / NB), 256, 0, stream>>>(lf, kw, sumsq);
  }

  k_tokstats  <<<CTOK / 8, 256, 0, stream>>>(lf, u, wv, mtok, dstok);
  k_scores    <<<CL, 256, 0, stream>>>(sumsq, mtok, dstok, scal, scores, conf);
  k_alloc     <<<1, 32, 0, stream>>>(conf, kplan);
  k_topk      <<<CL, 256, 0, stream>>>(scores, kplan, topv, topi, cand);
  k_select    <<<1, 256, 0, stream>>>(cand, topi, rowsel);
  k_gather    <<<64, 256, 0, stream>>>(lf, rowsel, out);
}
